// ParetoModel_10428180595349
// MI455X (gfx1250) — matrix operand order repaired and checked by reading, not compiled, not run on hardware
//
#include <hip/hip_runtime.h>
#include <hip/hip_bf16.h>
#include <stdint.h>

// Problem dims (fixed by reference)
constexpr int Bn = 4096, Ln = 128, Vn = 128, En = 64, Hn = 32, HIDn = 32, Gn = 128; // Gn = 4*H

typedef __attribute__((ext_vector_type(16))) _Float16 v16h;
typedef __attribute__((ext_vector_type(8)))  _Float16 v8h;
typedef __attribute__((ext_vector_type(8)))  float    v8f;
typedef __attribute__((ext_vector_type(4)))  float    v4f;

union HF { v16h v; unsigned u[8]; _Float16 f[16]; };

static __device__ __forceinline__ v8f wmma16(v16h a, v16h b, v8f c) {
  // D = A(16x32 f16) * B(32x16 f16) + C(16x16 f32)
  return __builtin_amdgcn_wmma_f32_16x16x32_f16(false, a, false, b, (short)0, c, false, false);
}

static __device__ __forceinline__ v16h cat16(v8h lo, v8h hi) {
  HF r;
  #pragma unroll
  for (int i = 0; i < 8; ++i) { r.f[i] = lo[i]; r.f[8 + i] = hi[i]; }
  return r.v;
}

// Fast transcendentals on the serial recurrence path:
//   v_exp_f32-based exp, v_rcp_f32-based reciprocal.
static __device__ __forceinline__ float fexp_(float x) {
  return __builtin_amdgcn_exp2f(x * 1.4426950408889634f);
}
static __device__ __forceinline__ float frcp_(float x) {
  return __builtin_amdgcn_rcpf(x);
}
static __device__ __forceinline__ float sigmoidf_(float x) {
  return frcp_(1.0f + fexp_(-x));
}
static __device__ __forceinline__ float tanhf_(float x) {
  // tanh(x) = 1 - 2/(exp(2x)+1); saturates correctly at +/-inf
  return 1.0f - 2.0f * frcp_(1.0f + fexp_(2.0f * x));
}

// Rebuild a B-matrix fragment (32x16, K = h index, N = batch) from per-lane h values.
// Lane holds h[batch=ln][j in {16p + r + 8*hl}] at f[8p+r].  B-frag element i <-> k = 16*hl + i.
static __device__ __forceinline__ HF mk_bfrag(const HF& own, int hl) {
  HF part, b;
  #pragma unroll
  for (int q = 0; q < 8; ++q)
    part.u[q] = (unsigned)__shfl_xor((int)own.u[q], 16, 32);
  #pragma unroll
  for (int q = 0; q < 4; ++q) {
    b.u[q]     = hl ? part.u[4 + q] : own.u[q];       // k 0..7  (hl=0) / 16..23 (hl=1)
    b.u[4 + q] = hl ? own.u[4 + q]  : part.u[q];      // k 8..15 (hl=0) / 24..31 (hl=1)
  }
  return b;
}

__global__ __launch_bounds__(128)
void lstm_head_kernel(
    const int* __restrict__ tokens, const int* __restrict__ next_idx,
    const float* __restrict__ emb,  const float* __restrict__ task,
    const float* __restrict__ history,
    const float* __restrict__ Wih0, const float* __restrict__ Whh0,
    const float* __restrict__ bih0, const float* __restrict__ bhh0,
    const float* __restrict__ Wih1, const float* __restrict__ Whh1,
    const float* __restrict__ bih1, const float* __restrict__ bhh1,
    const float* __restrict__ W1,   const float* __restrict__ b1,
    const float* __restrict__ W2,   const float* __restrict__ b2,
    float* __restrict__ out)
{
  __shared__ __align__(16) _Float16 sWh0[Gn * Hn];        // 128x32 f16 : W_hh0
  __shared__ __align__(16) _Float16 sW1c[Gn * (2 * Hn)];  // 128x64 f16 : [W_ih1 | W_hh1]
  __shared__ __align__(16) _Float16 sXG[Vn * Gn];         // 128x128 f16: emb@W_ih0.T + b0
  __shared__ __align__(16) float    sB1[Gn];              // b_ih1 + b_hh1

  const int tid  = threadIdx.x;
  const int lane = tid & 31;
  const int wv   = tid >> 5;      // wave in block (0..3)
  const int ln   = lane & 15;     // N column / batch-in-tile
  const int hl   = lane >> 4;     // half select

  // ---------------- one-time LDS fills ----------------
  for (int i = tid; i < Gn * Hn; i += 128) sWh0[i] = (_Float16)Whh0[i];
  for (int i = tid; i < Gn * 2 * Hn; i += 128) {
    int m = i >> 6, k = i & 63;
    sW1c[i] = (_Float16)((k < Hn) ? Wih1[m * Hn + k] : Whh1[m * Hn + (k - Hn)]);
  }
  if (tid < Gn) sB1[tid] = bih1[tid] + bhh1[tid];

  // ---------------- xg table via WMMA: sXG[v*128+g] ----------------
  // A = emb rows (M = vocab), B = W_ih0^T (N = gate). Wave wv does M-tiles 2wv, 2wv+1.
  // NOTE: nt loop deliberately NOT unrolled — keeps only one B-fragment's worth of
  // f32->f16 conversion registers live, so this one-time section does not set the
  // kernel-wide VGPR high-water mark (avoid >256 VGPRs / s_set_vgpr_msb path).
  #pragma unroll
  for (int mi = 0; mi < 2; ++mi) {
    const int mt = 2 * wv + mi;
    v16h afr[2];
    #pragma unroll
    for (int kc = 0; kc < 2; ++kc) {
      const float* p = emb + (16 * mt + ln) * En + 32 * kc + 16 * hl;
      v8h lo, hi;
      #pragma unroll
      for (int e = 0; e < 8; ++e) { lo[e] = (_Float16)p[e]; hi[e] = (_Float16)p[8 + e]; }
      afr[kc] = cat16(lo, hi);
    }
    #pragma unroll 1
    for (int nt = 0; nt < 8; ++nt) {
      v8f acc = {};
      #pragma unroll
      for (int kc = 0; kc < 2; ++kc) {
        const float* q = Wih0 + (16 * nt + ln) * En + 32 * kc + 16 * hl;
        HF bb;
        #pragma unroll
        for (int e = 0; e < 16; ++e) bb.f[e] = (_Float16)q[e];
        acc = wmma16(afr[kc], bb.v, acc);
      }
      const int g = 16 * nt + ln;
      const float bg = bih0[g] + bhh0[g];
      #pragma unroll
      for (int r = 0; r < 8; ++r) {
        const int v = 16 * mt + r + 8 * hl;
        sXG[v * Gn + g] = (_Float16)(acc[r] + bg);
      }
    }
  }
  __syncthreads();

  // ---------------- fused 2-layer LSTM recurrence ----------------
  const int row = blockIdx.x * 64 + wv * 16 + ln;   // batch row owned by this lane pair
  const int* tokrow = tokens + row * Ln;
  __builtin_prefetch(tokrow, 0, 3);                 // global_prefetch_b8 (pull token row)
  __builtin_prefetch(tokrow + 64, 0, 3);

  HF bh0, bh1;
  #pragma unroll
  for (int q = 0; q < 8; ++q) { bh0.u[q] = 0u; bh1.u[q] = 0u; }
  float c0s[2][8], c1s[2][8];
  #pragma unroll
  for (int p = 0; p < 2; ++p)
    #pragma unroll
    for (int r = 0; r < 8; ++r) { c0s[p][r] = 0.f; c1s[p][r] = 0.f; }

  int tok = tokrow[0];
  #pragma unroll 1
  for (int t = 0; t < Ln; ++t) {
    const int tok_next = (t + 1 < Ln) ? tokrow[t + 1] : 0;  // pipeline next token load
    v8f acc[8];
    // layer0: C init = xg table gather (bias already folded)
    #pragma unroll
    for (int mt = 0; mt < 8; ++mt) {
      v8h xv = *(const v8h*)(sXG + tok * Gn + 16 * mt + 8 * hl);
      #pragma unroll
      for (int r = 0; r < 8; ++r) acc[mt][r] = (float)xv[r];
    }
    // layer0 recurrent: g += W_hh0 * h0
    #pragma unroll
    for (int mt = 0; mt < 8; ++mt) {
      const _Float16* w = sWh0 + (16 * mt + ln) * Hn + 16 * hl;
      acc[mt] = wmma16(cat16(*(const v8h*)w, *(const v8h*)(w + 8)), bh0.v, acc[mt]);
    }
    HF own0;
    #pragma unroll
    for (int p = 0; p < 2; ++p)
      #pragma unroll
      for (int r = 0; r < 8; ++r) {
        float ig = sigmoidf_(acc[0 + p][r]);
        float fg = sigmoidf_(acc[2 + p][r]);
        float gg = tanhf_(acc[4 + p][r]);
        float og = sigmoidf_(acc[6 + p][r]);
        float c  = fg * c0s[p][r] + ig * gg;
        c0s[p][r] = c;
        own0.f[8 * p + r] = (_Float16)(og * tanhf_(c));
      }
    bh0 = mk_bfrag(own0, hl);

    // layer1: g = [W_ih1|W_hh1] * [h0;h1] + b1
    #pragma unroll
    for (int mt = 0; mt < 8; ++mt) {
      v4f blo = *(const v4f*)(sB1 + 16 * mt + 8 * hl);
      v4f bhi = *(const v4f*)(sB1 + 16 * mt + 8 * hl + 4);
      #pragma unroll
      for (int r = 0; r < 4; ++r) { acc[mt][r] = blo[r]; acc[mt][4 + r] = bhi[r]; }
      const _Float16* w = sW1c + (16 * mt + ln) * (2 * Hn) + 16 * hl;
      acc[mt] = wmma16(cat16(*(const v8h*)w,        *(const v8h*)(w + 8)), bh0.v, acc[mt]);
      acc[mt] = wmma16(cat16(*(const v8h*)(w + 32), *(const v8h*)(w + 40)), bh1.v, acc[mt]);
    }
    HF own1;
    #pragma unroll
    for (int p = 0; p < 2; ++p)
      #pragma unroll
      for (int r = 0; r < 8; ++r) {
        float ig = sigmoidf_(acc[0 + p][r]);
        float fg = sigmoidf_(acc[2 + p][r]);
        float gg = tanhf_(acc[4 + p][r]);
        float og = sigmoidf_(acc[6 + p][r]);
        float c  = fg * c1s[p][r] + ig * gg;
        c1s[p][r] = c;
        own1.f[8 * p + r] = (_Float16)(og * tanhf_(c));
      }
    bh1 = mk_bfrag(own1, hl);
    tok = tok_next;
  }

  // ---------------- head: relu(inp @ W1p.T + b1) @ W2.T + b2 ----------------
  // inp layout (K=160): [emb(64) | task(16)+pad(16) | h0(32) | h1(32)]
  const int nid = next_idx[row];
  v8f acc2[2];
  #pragma unroll
  for (int mt = 0; mt < 2; ++mt)
    #pragma unroll
    for (int r = 0; r < 8; ++r) acc2[mt][r] = b1[16 * mt + 8 * hl + r];

  v16h bk[5];
  #pragma unroll
  for (int kc = 0; kc < 2; ++kc) {
    const float* p = emb + nid * En + 32 * kc + 16 * hl;
    HF bb;
    #pragma unroll
    for (int e = 0; e < 16; ++e) bb.f[e] = (_Float16)p[e];
    bk[kc] = bb.v;
  }
  {
    HF bb;
    #pragma unroll
    for (int e = 0; e < 16; ++e) bb.f[e] = hl ? (_Float16)0.f : (_Float16)task[e];
    bk[2] = bb.v;
  }
  bk[3] = bh0.v;
  bk[4] = bh1.v;

  #pragma unroll
  for (int mt = 0; mt < 2; ++mt) {
    const int j = 16 * mt + ln;  // hidden unit (W1 row)
    #pragma unroll 1
    for (int kc = 0; kc < 5; ++kc) {
      HF aa;
      #pragma unroll
      for (int e = 0; e < 16; ++e) {
        const int kp = 32 * kc + 16 * hl + e;  // padded K index
        float w;
        if (kp < 80)       w = W1[j * 144 + kp];
        else if (kp < 96)  w = 0.f;                  // pad columns
        else               w = W1[j * 144 + kp - 16];
        aa.f[e] = (_Float16)w;
      }
      acc2[mt] = wmma16(aa.v, bk[kc], acc2[mt]);
    }
  }

  float s0 = 0.f, s1 = 0.f;
  #pragma unroll
  for (int mt = 0; mt < 2; ++mt)
    #pragma unroll
    for (int r = 0; r < 8; ++r) {
      const int j = 16 * mt + 8 * hl + r;
      const float hv = fmaxf(acc2[mt][r], 0.f);
      s0 += hv * W2[j];
      s1 += hv * W2[32 + j];
    }
  s0 += __shfl_xor(s0, 16, 32);
  s1 += __shfl_xor(s1, 16, 32);

  if (hl == 0) {
    const float sc0 = s0 + b2[0];
    const float sc1 = s1 + b2[1];
    const float hh0 = history[row * 3 + 0];
    const float hh1 = history[row * 3 + 1];
    const float cr  = history[row * 3 + 2];
    // real = hist2 + hist2*scores*flag, flag = [-1, 1]; normed = real*flag
    const float n0  = -(hh0 - hh0 * sc0);
    const float n1  =  (hh1 + hh1 * sc1);
    const float lcr = fmaxf(4.0f - cr, 0.0f) + 1.0f;
    out[row * 2 + 0] = sc0;
    out[row * 2 + 1] = sc1;
    out[2 * Bn + row * 2 + 0] = n0 / lcr;
    out[2 * Bn + row * 2 + 1] = n1 / lcr;
  }
}

// pareto_mask[i] = !exists j: adj[j] strictly dominates adj[i] in both components
__global__ __launch_bounds__(256)
void pareto_kernel(const float* __restrict__ adj, float* __restrict__ mask) {
  __shared__ float2 tile[256];
  const int i = blockIdx.x * 256 + threadIdx.x;
  const float a0 = adj[2 * i];
  const float a1 = adj[2 * i + 1];
  int dom = 0;
  for (int base = 0; base < Bn; base += 256) {
    __syncthreads();
    tile[threadIdx.x] = ((const float2*)adj)[base + threadIdx.x];
    __syncthreads();
    #pragma unroll 8
    for (int j = 0; j < 256; ++j) {
      const float2 b = tile[j];
      dom |= (int)((b.x > a0) & (b.y > a1));
    }
  }
  mask[i] = dom ? 0.0f : 1.0f;
}

extern "C" void kernel_launch(void* const* d_in, const int* in_sizes, int n_in,
                              void* d_out, int out_size, void* d_ws, size_t ws_size,
                              hipStream_t stream) {
  (void)in_sizes; (void)n_in; (void)out_size; (void)d_ws; (void)ws_size;
  const int*   tokens = (const int*)d_in[0];
  const int*   nidx   = (const int*)d_in[1];
  const float* emb    = (const float*)d_in[2];
  const float* task   = (const float*)d_in[3];
  const float* hist   = (const float*)d_in[4];
  const float* Wih0   = (const float*)d_in[5];
  const float* Whh0   = (const float*)d_in[6];
  const float* bih0   = (const float*)d_in[7];
  const float* bhh0   = (const float*)d_in[8];
  const float* Wih1   = (const float*)d_in[9];
  const float* Whh1   = (const float*)d_in[10];
  const float* bih1   = (const float*)d_in[11];
  const float* bhh1   = (const float*)d_in[12];
  const float* W1     = (const float*)d_in[13];
  const float* b1     = (const float*)d_in[14];
  const float* W2     = (const float*)d_in[15];
  const float* b2     = (const float*)d_in[16];
  float* out = (float*)d_out;

  lstm_head_kernel<<<Bn / 64, 128, 0, stream>>>(
      tokens, nidx, emb, task, hist,
      Wih0, Whh0, bih0, bhh0, Wih1, Whh1, bih1, bhh1,
      W1, b1, W2, b2, out);

  pareto_kernel<<<Bn / 256, 256, 0, stream>>>(out + 2 * Bn, out + 4 * Bn);
}
